// CausalSelfAttention_38946763440480
// MI455X (gfx1250) — compile-verified, matrix operand order checked by reading, not run on hardware
//
#include <hip/hip_runtime.h>

typedef __attribute__((ext_vector_type(16))) __bf16 v16bf;
typedef __attribute__((ext_vector_type(8)))  float  v8f;
typedef __attribute__((ext_vector_type(4)))  int    v4i;

#define B_   2
#define T_   2048
#define C_   2048
#define NH   32
#define NG   8
#define QPK  4
#define HS   64
#define NQKV 3072   // (NH + 2*NG) * HS

#define AS1 __attribute__((address_space(1)))
#define AS3 __attribute__((address_space(3)))

static __device__ inline v8f vzero8() {
  v8f z = {0.f,0.f,0.f,0.f,0.f,0.f,0.f,0.f};
  return z;
}

// 16-byte global -> LDS copy; async (ASYNCcnt-tracked, no VGPR bounce) when available.
static __device__ inline void lds_cp16(void* lds_dst, const void* gsrc) {
#if __has_builtin(__builtin_amdgcn_global_load_async_to_lds_b128)
  __builtin_amdgcn_global_load_async_to_lds_b128(
      (AS1 v4i*)gsrc, (AS3 v4i*)lds_dst, 0, 0);
#else
  *(int4*)lds_dst = *(const int4*)gsrc;
#endif
}

static __device__ inline void wait_async_lds() {
#if __has_builtin(__builtin_amdgcn_global_load_async_to_lds_b128)
#if __has_builtin(__builtin_amdgcn_s_wait_asynccnt)
  __builtin_amdgcn_s_wait_asynccnt(0);
#else
  asm volatile("s_wait_asynccnt 0x0" ::: "memory");
#endif
#endif
}

// ---------------- f32 -> bf16 conversion ----------------
__global__ void cvt_bf16_kernel(const float* __restrict__ in,
                                __bf16* __restrict__ out, int n) {
  int i = blockIdx.x * 256 + threadIdx.x;
  if (i < n) out[i] = (__bf16)in[i];
}

// ---------------- WMMA GEMM: out[M,N] = A[M,K] @ W[N,K]^T + bias ----------------
// Block tile 128x256 (8 waves as 2x4, each wave 64x64), K stepped by 32.
// A(128x32) + B(256x32) staged in LDS via async copies each K-step.
__global__ __launch_bounds__(256) void gemm_bf16_wmma(
    const __bf16* __restrict__ A, const __bf16* __restrict__ W,
    const float* __restrict__ bias, float* __restrict__ out,
    int M, int N, int K)
{
  __shared__ __bf16 As[128 * 32];   //  8 KB
  __shared__ __bf16 Bs[256 * 32];   // 16 KB

  const int tid  = threadIdx.x;
  const int lane = tid & 31;
  const int l16  = lane & 15;
  const int half = lane >> 4;
  const int wave = tid >> 5;            // 0..7
  const int wr   = wave >> 2;           // 0..1  (wave row)
  const int wc   = wave & 3;            // 0..3  (wave col)
  const int m_off = wr * 64;
  const int n_off = wc * 64;

  const int tilesN = N >> 8;            // N / 256
  const int bm = blockIdx.x / tilesN;
  const int bn = blockIdx.x % tilesN;
  const int m0 = bm * 128;
  const int n0 = bn * 256;

  v8f acc[4][4];
#pragma unroll
  for (int i = 0; i < 4; i++)
#pragma unroll
    for (int j = 0; j < 4; j++) acc[i][j] = vzero8();

  for (int kk = 0; kk < K; kk += 32) {
    // ---- stage A tile: 128 rows x 32 cols = 512 x 16B chunks, 2 per thread ----
#pragma unroll
    for (int it = 0; it < 2; it++) {
      int ch = tid + it * 256;
      int r  = ch >> 2;
      int cc = (ch & 3) * 8;
      lds_cp16(As + r * 32 + cc, A + (size_t)(m0 + r) * K + kk + cc);
    }
    // ---- stage B tile: 256 rows x 32 cols = 1024 x 16B chunks, 4 per thread ----
#pragma unroll
    for (int it = 0; it < 4; it++) {
      int ch = tid + it * 256;
      int r  = ch >> 2;
      int cc = (ch & 3) * 8;
      lds_cp16(Bs + r * 32 + cc, W + (size_t)(n0 + r) * K + kk + cc);
    }
    wait_async_lds();
    __syncthreads();

    // ---- 16 WMMAs per wave from LDS ----
    v16bf va[4], vb[4];
#pragma unroll
    for (int i = 0; i < 4; i++)
      va[i] = *(const v16bf*)(As + (m_off + i*16 + l16) * 32 + half*16);
#pragma unroll
    for (int j = 0; j < 4; j++)
      vb[j] = *(const v16bf*)(Bs + (n_off + j*16 + l16) * 32 + half*16);
#pragma unroll
    for (int i = 0; i < 4; i++)
#pragma unroll
      for (int j = 0; j < 4; j++)
        acc[i][j] = __builtin_amdgcn_wmma_f32_16x16x32_bf16(
            false, va[i], false, vb[j], (short)0, acc[i][j], false, false);

    __syncthreads();
  }

#pragma unroll
  for (int i = 0; i < 4; i++) {
#pragma unroll
    for (int j = 0; j < 4; j++) {
      int col = n0 + n_off + j*16 + l16;
      float bcol = bias[col];
#pragma unroll
      for (int r = 0; r < 8; r++) {
        int row = m0 + m_off + i*16 + half*8 + r;
        out[(size_t)row * N + col] = acc[i][j][r] + bcol;
      }
    }
  }
}

// ---------------- RoPE + split into q/k (rotated) and V^T ----------------
__global__ void rope_split_kernel(const float* __restrict__ qkv,
                                  const float* __restrict__ cosp,
                                  const float* __restrict__ sinp,
                                  __bf16* __restrict__ qb,
                                  __bf16* __restrict__ kb,
                                  __bf16* __restrict__ vbt)
{
  int idx = blockIdx.x * 256 + threadIdx.x;
  const int total = B_ * T_ * NQKV;
  if (idx >= total) return;
  int n  = idx % NQKV;
  int bt = idx / NQKV;
  int t  = bt % T_;
  int b  = bt / T_;
  int g    = n / (6 * HS);
  int rem  = n % (6 * HS);
  int slot = rem / HS;
  int d    = rem % HS;
  float x = qkv[idx];

  if (slot == 5) { // V passthrough, stored transposed [g][d][s]
    vbt[((size_t)(b*NG + g) * HS + d) * T_ + t] = (__bf16)x;
    return;
  }
  // RoPE for q (slot 0..3) and k (slot 4)
  float c = cosp[t * HS + d];
  float s = sinp[t * HS + d];
  int   pd    = (d < HS/2) ? d + HS/2 : d - HS/2;
  float other = qkv[(size_t)bt * NQKV + (n - d) + pd];
  float rot   = (d < HS/2) ? -other : other;
  float val   = x * c + rot * s;
  if (slot < QPK) {
    val *= 0.125f; // 1/sqrt(HS) folded into Q
    qb[(((size_t)(b*NG + g) * QPK + slot) * T_ + t) * HS + d] = (__bf16)val;
  } else {
    kb[((size_t)(b*NG + g) * T_ + t) * HS + d] = (__bf16)val;
  }
}

// ---------------- Flash attention: 1 wave = one (b, head, 16-row Q tile) ----------------
__global__ __launch_bounds__(128) void attn_kernel(
    const __bf16* __restrict__ qb, const __bf16* __restrict__ kb,
    const __bf16* __restrict__ vbt, __bf16* __restrict__ yb)
{
  __shared__ __bf16 lds_p[4][16 * 64];
  const int lane = threadIdx.x & 31;
  const int l16  = lane & 15;
  const int half = lane >> 4;
  const int wib  = threadIdx.x >> 5;

  int w  = blockIdx.x * 4 + wib;
  int mt = w % (T_ / 16);
  int h  = (w / (T_ / 16)) % NH;
  int b  = w / ((T_ / 16) * NH);
  int g  = h / QPK, qi = h % QPK;

  const __bf16* qhead  = qb  + ((size_t)(b*NG + g) * QPK + qi) * T_ * HS;
  const __bf16* khead  = kb  + (size_t)(b*NG + g) * T_ * HS;
  const __bf16* vthead = vbt + (size_t)(b*NG + g) * HS * T_;
  int m0 = mt * 16;

  v16bf qa[2];
#pragma unroll
  for (int h2 = 0; h2 < 2; h2++)
    qa[h2] = *(const v16bf*)(qhead + (size_t)(m0 + l16) * HS + h2*32 + half*16);

  v8f ya[4];
#pragma unroll
  for (int j = 0; j < 4; j++) ya[j] = vzero8();
  float rm[8], rs[8];
#pragma unroll
  for (int r = 0; r < 8; r++) { rm[r] = -1e30f; rs[r] = 0.f; }

  __bf16* pt = lds_p[wib];
  const int send = m0 + 16;

  for (int s0 = 0; s0 < send; s0 += 64) {
    // ---- S tile = Q @ K^T (16 x 64) ----
    v8f sc[4];
#pragma unroll
    for (int j = 0; j < 4; j++) sc[j] = vzero8();
#pragma unroll
    for (int j = 0; j < 4; j++)
#pragma unroll
      for (int h2 = 0; h2 < 2; h2++) {
        v16bf kop = *(const v16bf*)(khead + (size_t)(s0 + j*16 + l16) * HS + h2*32 + half*16);
        sc[j] = __builtin_amdgcn_wmma_f32_16x16x32_bf16(
            false, qa[h2], false, kop, (short)0, sc[j], false, false);
      }

    // ---- causal mask + chunk row-max (D layout: lane=col, vgpr r=row half*8+r) ----
    float cm[8];
#pragma unroll
    for (int r = 0; r < 8; r++) cm[r] = -1e30f;
#pragma unroll
    for (int j = 0; j < 4; j++) {
      int scol = s0 + j*16 + l16;
#pragma unroll
      for (int r = 0; r < 8; r++) {
        int grow = m0 + half*8 + r;
        float v = sc[j][r];
        if (scol > grow) v = -1e30f;
        sc[j][r] = v;
        cm[r] = fmaxf(cm[r], v);
      }
    }
#pragma unroll
    for (int m = 1; m < 16; m <<= 1)
#pragma unroll
      for (int r = 0; r < 8; r++) cm[r] = fmaxf(cm[r], __shfl_xor(cm[r], m, 32));

    // ---- online softmax rescale ----
    float nm[8], al[8];
#pragma unroll
    for (int r = 0; r < 8; r++) {
      nm[r] = fmaxf(rm[r], cm[r]);
      al[r] = __expf(rm[r] - nm[r]);
      rm[r] = nm[r];
      rs[r] *= al[r];
    }
#pragma unroll
    for (int j = 0; j < 4; j++)
#pragma unroll
      for (int r = 0; r < 8; r++) ya[j][r] *= al[r];

    // ---- P = exp(S - max), partial row-sums, stage P (bf16) to LDS ----
#pragma unroll
    for (int j = 0; j < 4; j++)
#pragma unroll
      for (int r = 0; r < 8; r++) {
        float p = __expf(sc[j][r] - nm[r]);
        rs[r] += p;
        pt[(half*8 + r) * 64 + j*16 + l16] = (__bf16)p;
      }
    asm volatile("s_wait_dscnt 0x0" ::: "memory");

    // ---- reload P in A layout, y += P @ V ----
    v16bf pa[2];
#pragma unroll
    for (int h2 = 0; h2 < 2; h2++)
      pa[h2] = *(const v16bf*)(pt + l16*64 + h2*32 + half*16);
#pragma unroll
    for (int jn = 0; jn < 4; jn++)
#pragma unroll
      for (int h2 = 0; h2 < 2; h2++) {
        v16bf bv = *(const v16bf*)(vthead + (size_t)(jn*16 + l16) * T_ + s0 + h2*32 + half*16);
        ya[jn] = __builtin_amdgcn_wmma_f32_16x16x32_bf16(
            false, pa[h2], false, bv, (short)0, ya[jn], false, false);
      }
  }

  // ---- final row-sum reduce + normalize + write [b, t, h*64+d] as bf16 ----
#pragma unroll
  for (int m = 1; m < 16; m <<= 1)
#pragma unroll
    for (int r = 0; r < 8; r++) rs[r] += __shfl_xor(rs[r], m, 32);
  float inv[8];
#pragma unroll
  for (int r = 0; r < 8; r++) inv[r] = 1.f / rs[r];
#pragma unroll
  for (int jn = 0; jn < 4; jn++) {
    int d = jn*16 + l16;
#pragma unroll
    for (int r = 0; r < 8; r++) {
      int t = m0 + half*8 + r;
      yb[((size_t)b * T_ + t) * (NH*HS) + h*HS + d] = (__bf16)(ya[jn][r] * inv[r]);
    }
  }
}

// ---------------- host launcher ----------------
extern "C" void kernel_launch(void* const* d_in, const int* in_sizes, int n_in,
                              void* d_out, int out_size, void* d_ws, size_t ws_size,
                              hipStream_t stream) {
  const float* x      = (const float*)d_in[0];
  const float* cosp   = (const float*)d_in[1];
  const float* sinp   = (const float*)d_in[2];
  const float* attn_w = (const float*)d_in[3];
  const float* attn_b = (const float*)d_in[4];
  const float* proj_w = (const float*)d_in[5];
  const float* proj_b = (const float*)d_in[6];

  char* ws = (char*)d_ws;
  // workspace layout (bytes)
  __bf16* xb    = (__bf16*)(ws);                       // 4096*2048*2   = 16 MB
  __bf16* wqkv  = (__bf16*)(ws + 16777216);            // 3072*2048*2   = 12 MB
  __bf16* wproj = (__bf16*)(ws + 29360128);            // 2048*2048*2   =  8 MB
  float*  qkv   = (float*)(ws + 37748736);             // 4096*3072*4   = 48 MB
  __bf16* qb    = (__bf16*)(ws + 88080384);            // 2*8*4*2048*64 = 16 MB
  __bf16* kb    = (__bf16*)(ws + 104857600);           // 2*8*2048*64   =  4 MB
  __bf16* vbt   = (__bf16*)(ws + 109051904);           // 2*8*64*2048   =  4 MB
  __bf16* yb    = (__bf16*)(ws + 113246208);           // 4096*2048*2   = 16 MB

  const int nX  = B_ * T_ * C_;          // 8388608
  const int nWq = NQKV * C_;             // 6291456
  const int nWp = C_ * (NH * HS);        // 4194304

  cvt_bf16_kernel<<<(nX  + 255) / 256, 256, 0, stream>>>(x,      xb,    nX);
  cvt_bf16_kernel<<<(nWq + 255) / 256, 256, 0, stream>>>(attn_w, wqkv,  nWq);
  cvt_bf16_kernel<<<(nWp + 255) / 256, 256, 0, stream>>>(proj_w, wproj, nWp);

  // QKV GEMM: [4096,3072] = [4096,2048] @ [3072,2048]^T ; block tile 128x256
  gemm_bf16_wmma<<<(B_*T_/128) * (NQKV/256), 256, 0, stream>>>(
      xb, wqkv, attn_b, qkv, B_*T_, NQKV, C_);

  const int nR = B_ * T_ * NQKV;         // 12582912
  rope_split_kernel<<<(nR + 255) / 256, 256, 0, stream>>>(qkv, cosp, sinp, qb, kb, vbt);

  // attention: B*NH*(T/16) = 8192 waves, 4 waves per block
  attn_kernel<<<B_ * NH * (T_/16) / 4, 128, 0, stream>>>(qb, kb, vbt, yb);

  // projection: [4096,2048] = [4096,2048] @ [2048,2048]^T ; block tile 128x256
  gemm_bf16_wmma<<<(B_*T_/128) * (C_/256), 256, 0, stream>>>(
      yb, wproj, proj_b, (float*)d_out, B_*T_, C_, C_);
}
